// ParallelScopeSum_23527830847520
// MI455X (gfx1250) — matrix operand order repaired and checked by reading, not compiled, not run on hardware
//
#include <hip/hip_runtime.h>
#include <hip/hip_bf16.h>

// Problem shape (fixed by the reference)
#define SD   256      // S*D batches
#define BB   256      // rows (b)
#define NN   512      // reduction dim (n)
#define KK   512      // cols (k)

typedef __attribute__((ext_vector_type(16))) _Float16 v16h;
typedef __attribute__((ext_vector_type(8)))  _Float16 v8h;
typedef __attribute__((ext_vector_type(8)))  float    v8f;

// fast hardware log/exp (values here are never denormal; see analysis)
#if __has_builtin(__builtin_amdgcn_logf)
#define FAST_LN(x)  (__builtin_amdgcn_logf(x) * 0.69314718055994531f)
#else
#define FAST_LN(x)  __logf(x)
#endif
#if __has_builtin(__builtin_amdgcn_exp2f)
#define FAST_EXP(x) __builtin_amdgcn_exp2f((x) * 1.44269504088896341f)
#else
#define FAST_EXP(x) __expf(x)
#endif

// CDNA5 async copy: global -> LDS (16B per lane), tracked by ASYNCcnt.
__device__ __forceinline__ void async_copy_b128(unsigned lds_off,
                                                const void* gaddr) {
  asm volatile("global_load_async_to_lds_b128 %0, %1, off"
               :: "v"(lds_off), "v"((unsigned long long)(uintptr_t)gaddr)
               : "memory");
}
__device__ __forceinline__ unsigned lds_off32(const void* p) {
  return (unsigned)(uintptr_t)p;   // aperture in [63:32]; low 32 = LDS offset
}

// ---------------------------------------------------------------------------
// Kernel 1: per-(sd,k) column stats over n of acc; write
//   wbias[sd*KK + k] = log(colmax) - log(colsum)
//   Wt[(sd*KK + k)*NN + n] = (f16)(acc[sd][n][k] / colmax)   (transposed!)
// ---------------------------------------------------------------------------
__global__ void colstats_kernel(const float* __restrict__ acc,
                                _Float16* __restrict__ Wt,
                                float* __restrict__ wbias) {
  const int k  = blockIdx.x * 256 + threadIdx.x;
  const int sd = blockIdx.y;
  const float* col = acc + (size_t)sd * NN * KK + k;   // stride KK over n

  float cmax = -1.0f, csum = 0.0f;
#pragma unroll 8
  for (int n = 0; n < NN; ++n) {
    float v = col[(size_t)n * KK];
    cmax = fmaxf(cmax, v);
    csum += v;
  }
  wbias[(size_t)sd * KK + k] = FAST_LN(cmax) - FAST_LN(csum);

  const float inv = __frcp_rn(cmax);
  _Float16* wrow = Wt + ((size_t)sd * KK + k) * NN;    // contiguous over n
#pragma unroll 8
  for (int n = 0; n < NN; ++n) {
    float v = col[(size_t)n * KK];
    wrow[n] = (_Float16)(v * inv);
  }
}

// ---------------------------------------------------------------------------
// Kernel 2: per-(sd,b) row max over n of x; one wave32 per row.
//   xmax[row] = max_n x ;  Ah[row*NN + n] = (f16)exp(x - max)
// ---------------------------------------------------------------------------
__global__ void rowstats_kernel(const float* __restrict__ x,
                                _Float16* __restrict__ Ah,
                                float* __restrict__ xmax) {
  const int lane = threadIdx.x & 31;
  const int wave = threadIdx.x >> 5;
  const int row  = blockIdx.x * 8 + wave;              // flat (sd*BB + b)
  const float* xr = x + (size_t)row * NN;

  float v[16];
  float m = -3.4e38f;
#pragma unroll
  for (int i = 0; i < 16; ++i) {
    v[i] = xr[lane + i * 32];
    m = fmaxf(m, v[i]);
  }
#pragma unroll
  for (int off = 16; off >= 1; off >>= 1)
    m = fmaxf(m, __shfl_xor(m, off, 32));

  _Float16* ar = Ah + (size_t)row * NN;
#pragma unroll
  for (int i = 0; i < 16; ++i)
    ar[lane + i * 32] = (_Float16)FAST_EXP(v[i] - m);

  if (lane == 0) xmax[row] = m;
}

// ---------------------------------------------------------------------------
// Kernel 3: per-batch GEMM  C[b,k] = Ah[b,:] . Wt[k,:]  (both f16, contiguous
// over reduction dim n), epilogue out = log(C) + xmax[b] + wbias[k].
//
// grid (KK/128, BB/128, SD); block 256 = 8 waves (2 x 4 wave grid).
// Block tile: 128x128.  Wave tile: 64x32 = 4x2 WMMA 16x16 accumulators.
// k-step = 32; DOUBLE-buffered LDS filled by async-to-LDS DMA (ASYNCcnt).
// ---------------------------------------------------------------------------
#define LDS_STRIDE 40
#define TILE_H     (128 * LDS_STRIDE)

__global__ __launch_bounds__(256)
void logmm_wmma_kernel(const _Float16* __restrict__ Ah,
                       const _Float16* __restrict__ Wt,
                       const float* __restrict__ xmax,
                       const float* __restrict__ wbias,
                       float* __restrict__ out) {
  __shared__ __align__(16) _Float16 sA[2][TILE_H];
  __shared__ __align__(16) _Float16 sW[2][TILE_H];

  const int tid  = threadIdx.x;
  const int lane = tid & 31;
  const int wave = tid >> 5;
  const int sd   = blockIdx.z;
  const int k0   = blockIdx.x * 128;                  // output col tile base
  const int b0   = blockIdx.y * 128;                  // output row tile base
  const int wmb  = (wave >> 2) * 64;                  // wave row base in tile
  const int wnb  = (wave & 3) * 32;                   // wave col base in tile

  // staging assignment: 2 threads per 32-half row; each moves 2x16B per tile
  const int ldr = tid >> 1;                           // 0..127
  const int ldc = (tid & 1) * 16;                     // 0 or 16

  const _Float16* gA = Ah + ((size_t)sd * BB + b0 + ldr) * NN + ldc;
  const _Float16* gW = Wt + ((size_t)sd * KK + k0 + ldr) * NN + ldc;
  const unsigned lAo[2] = { lds_off32(&sA[0][ldr * LDS_STRIDE + ldc]),
                            lds_off32(&sA[1][ldr * LDS_STRIDE + ldc]) };
  const unsigned lWo[2] = { lds_off32(&sW[0][ldr * LDS_STRIDE + ldc]),
                            lds_off32(&sW[1][ldr * LDS_STRIDE + ldc]) };

  v8f accf[4][2] = {};

  const int rsel = (lane < 16) ? 0 : 8;               // A frag column select
  const int bsel = (lane < 16) ? 0 : 16;              // B frag column select
  const int lm   = lane & 15;

  // issue tile `step` (4 async b128 per thread) into buffer step&1
  auto issue = [&](int step) {
    const int buf = step & 1;
    const _Float16* pa = gA + step * 32;
    const _Float16* pw = gW + step * 32;
    async_copy_b128(lAo[buf],      pa);
    async_copy_b128(lAo[buf] + 16, pa + 8);
    async_copy_b128(lWo[buf],      pw);
    async_copy_b128(lWo[buf] + 16, pw + 8);
  };

  issue(0);
  for (int step = 0; step < 16; ++step) {
    const int buf = step & 1;
    if (step + 1 < 16) {
      issue(step + 1);                                // prefetch next tile
      asm volatile("s_wait_asynccnt 0x4" ::: "memory");  // current tile landed
    } else {
      asm volatile("s_wait_asynccnt 0x0" ::: "memory");
    }
    __syncthreads();                                  // tile visible to all

    // A fragments: lane<16 holds M=lm, K in {0..7,16..23}; lane>=16 shifts +8
    v16h afrag[4];
#pragma unroll
    for (int f = 0; f < 4; ++f) {
      const _Float16* r = &sA[buf][(wmb + f * 16 + lm) * LDS_STRIDE];
      v8h lo = *(const v8h*)(r + rsel);
      v8h hi = *(const v8h*)(r + rsel + 16);
      afrag[f] = __builtin_shufflevector(lo, hi,
                   0,1,2,3,4,5,6,7,8,9,10,11,12,13,14,15);
    }
    // B fragments: lane = column k; lane<16 holds K(n) in {0..7,16..23}; lane>=16 shifts +8
    v16h bfrag[2];
#pragma unroll
    for (int j = 0; j < 2; ++j) {
      const _Float16* r = &sW[buf][(wnb + j * 16 + lm) * LDS_STRIDE];
      v8h lo = *(const v8h*)(r + rsel);
      v8h hi = *(const v8h*)(r + rsel + 16);
      bfrag[j] = __builtin_shufflevector(lo, hi,
                   0,1,2,3,4,5,6,7,8,9,10,11,12,13,14,15);
    }

#pragma unroll
    for (int f = 0; f < 4; ++f)
#pragma unroll
      for (int j = 0; j < 2; ++j)
        accf[f][j] = __builtin_amdgcn_wmma_f32_16x16x32_f16(
            false, afrag[f], false, bfrag[j],
            (short)0, accf[f][j], false, false);

    __syncthreads();   // all reads of this buffer done before its re-fill
  }

  // epilogue: out[b,k] = log(acc) + xmax[b] + wbias[k]
  float wb[2];
#pragma unroll
  for (int j = 0; j < 2; ++j)
    wb[j] = wbias[(size_t)sd * KK + k0 + wnb + j * 16 + lm];

#pragma unroll
  for (int f = 0; f < 4; ++f) {
#pragma unroll
    for (int r = 0; r < 8; ++r) {
      const int b = b0 + wmb + f * 16 + r + rsel;     // D layout: VGPR r -> M
      const float xm = xmax[(size_t)sd * BB + b];
#pragma unroll
      for (int j = 0; j < 2; ++j) {
        const int k = k0 + wnb + j * 16 + lm;
        out[((size_t)sd * BB + b) * KK + k] =
            FAST_LN(accf[f][j][r]) + xm + wb[j];
      }
    }
  }
}

// ---------------------------------------------------------------------------
extern "C" void kernel_launch(void* const* d_in, const int* in_sizes, int n_in,
                              void* d_out, int out_size, void* d_ws, size_t ws_size,
                              hipStream_t stream) {
  (void)in_sizes; (void)n_in; (void)out_size; (void)ws_size;
  const float* x   = (const float*)d_in[0];            // (S,D,B,N) f32
  const float* acc = (const float*)d_in[1];            // (S,D,N,K) f32
  float* out = (float*)d_out;                          // (S,D,B,K) f32

  char* ws = (char*)d_ws;
  _Float16* Wt   = (_Float16*)(ws);                                   // 134.2 MB
  _Float16* Ah   = (_Float16*)(ws + (size_t)SD * KK * NN * 2);        //  67.1 MB
  float*    xmax = (float*)(ws + (size_t)SD * KK * NN * 2
                               + (size_t)SD * BB * NN * 2);           // 256 KB
  float*    wbias= (float*)((char*)xmax + (size_t)SD * BB * 4);       // 512 KB

  colstats_kernel<<<dim3(KK / 256, SD), 256, 0, stream>>>(acc, Wt, wbias);
  rowstats_kernel<<<dim3(SD * BB / 8), 256, 0, stream>>>(x, Ah, xmax);
  logmm_wmma_kernel<<<dim3(KK / 128, BB / 128, SD), 256, 0, stream>>>(
      Ah, Wt, xmax, wbias, out);
}
